// TCN2d_inf_iterative_pruning_12515534701277
// MI455X (gfx1250) — 8-bit matrix operand order repaired and checked by reading, compiled, not run on hardware
//
#include <hip/hip_runtime.h>
#include <hip/hip_bf16.h>

typedef int v8i __attribute__((ext_vector_type(8)));

#define BN_EPS 1e-5f

__device__ __forceinline__ int clampi(int v, int lo, int hi) {
    return v < lo ? lo : (v > hi ? hi : v);
}
__device__ __forceinline__ int q_rne(float x) { return (int)rintf(x); }

__global__ void pack_weights_kernel(const float* __restrict__ w,
                                    const float* __restrict__ wsc, int wi,
                                    int Cin, int KH, int Cout,
                                    int ksteps, int mtiles,
                                    int* __restrict__ outp)
{
    int total = ksteps * mtiles * 32 * 8;
    int id = blockIdx.x * blockDim.x + threadIdx.x;
    if (id >= total) return;
    float inv = 1.0f / wsc[wi];
    int v    = id & 7;
    int rest = id >> 3;
    int lane = rest & 31;
    rest >>= 5;
    int mt = rest % mtiles;
    int ks = rest / mtiles;
    int half = lane >> 4;
    int co   = mt * 16 + (lane & 15);
    unsigned dw = 0;
    for (int j = 0; j < 4; ++j) {
        int Klocal = ((v >> 2) * 32) + (half * 16) + ((v & 3) * 4) + j;
        int Kg = ks * 64 + Klocal;
        int kk = Kg / Cin;
        int ci = Kg - kk * Cin;
        int q = 0;
        if (co < Cout && kk < KH) {
            float val = w[(size_t)(co * Cin + ci) * KH + kk];
            q = clampi(q_rne(val * inv), -128, 127);
        }
        dw |= ((unsigned)(q & 0xff)) << (8 * j);
    }
    outp[id] = (int)dw;
}

__global__ void prep_kernel(const float* __restrict__ x,
                            const int* __restrict__ idx_in,
                            const float* __restrict__ act_scales,
                            signed char* __restrict__ out0)
{
    const int T = 12288, Cin = 64, Tfull = 16384;
    size_t id = (size_t)blockIdx.x * blockDim.x + threadIdx.x;
    size_t total = (size_t)32 * T * Cin;
    if (id >= total) return;
    int ci = (int)(id & 63);
    int t  = (int)((id >> 6) % T);
    int b  = (int)(id / ((size_t)T * Cin));
    float s0 = act_scales[0], s1 = act_scales[1];
    float v = x[((size_t)(b * Cin + ci)) * Tfull + idx_in[t]];
    int q0 = clampi(q_rne(v / s0), -128, 127);
    float v0 = (float)q0 * s0;
    int q1 = clampi(q_rne(v0 / s1), -128, 127);
    out0[id] = (signed char)q1;
}

__launch_bounds__(256, 2)
__global__ void conv_i8_wmma_kernel(const signed char* __restrict__ in, int Tbuf,
                                    const int* __restrict__ gather, int Tg,
                                    int Cin, int ksteps,
                                    const int* __restrict__ wq,
                                    const float* __restrict__ act_scales,
                                    int si_in, int si_relu, int si_next,
                                    const float* __restrict__ w_scales, int wi,
                                    const float* __restrict__ bn, int bnoff,
                                    signed char* __restrict__ out, int Tout)
{
    __shared__ __align__(16) signed char lds_act[132 * 128];
    const int tid  = threadIdx.x;
    const int lane = tid & 31;
    const int wave = tid >> 5;
    const int b    = blockIdx.y;
    const int t0   = blockIdx.x * 128;

    const int cvec  = Cin >> 4;
    const int units = 130 * cvec;
    for (int u = tid; u < units; u += 256) {
        int row = u / cvec;
        int off = u - row * cvec;
        int t = t0 + row; if (t >= Tg) t = Tg - 1;
        int tg = gather ? gather[t] : t;
        const int4* src = (const int4*)(in + ((size_t)b * Tbuf + tg) * Cin) + off;
        *((int4*)(lds_act + row * Cin) + off) = *src;
    }
    __syncthreads();

    const int n    = lane & 15;
    const int half = lane >> 4;

    v8i acc[8];
#pragma unroll
    for (int nt = 0; nt < 8; ++nt) acc[nt] = v8i{};

    for (int ks = 0; ks < ksteps; ++ks) {
        const int4* wp = (const int4*)wq + ((((ks * 8) + wave) * 32 + lane) << 1);
        int4 a0 = wp[0], a1 = wp[1];
        v8i a;
        a[0] = a0.x; a[1] = a0.y; a[2] = a0.z; a[3] = a0.w;
        a[4] = a1.x; a[5] = a1.y; a[6] = a1.z; a[7] = a1.w;
        int k   = (ks * 64) / Cin;
        int cib = (ks * 64) - k * Cin;
#pragma unroll
        for (int nt = 0; nt < 8; ++nt) {
            const signed char* bp =
                lds_act + (size_t)(nt * 16 + n + k) * Cin + cib + half * 16;
            int4 blo = *(const int4*)bp;
            int4 bhi = *(const int4*)(bp + 32);
            v8i bf;
            bf[0] = blo.x; bf[1] = blo.y; bf[2] = blo.z; bf[3] = blo.w;
            bf[4] = bhi.x; bf[5] = bhi.y; bf[6] = bhi.z; bf[7] = bhi.w;
            acc[nt] = __builtin_amdgcn_wmma_i32_16x16x64_iu8(
                true, a, true, bf, acc[nt], false, false);
        }
    }

    float deq    = act_scales[si_in] * w_scales[wi];
    float s_next = act_scales[si_next];
    float s_relu = (si_relu >= 0) ? act_scales[si_relu] : 1.0f;

    float Ar[8], Cr[8];
#pragma unroll
    for (int r = 0; r < 8; ++r) {
        int co  = wave * 16 + half * 8 + r;
        float g  = bn[bnoff + 0 * 128 + co];
        float bb = bn[bnoff + 1 * 128 + co];
        float mm = bn[bnoff + 2 * 128 + co];
        float vv = bn[bnoff + 3 * 128 + co];
        float sc = g * rsqrtf(vv + BN_EPS);
        Ar[r] = deq * sc;
        Cr[r] = bb - mm * sc;
    }

#pragma unroll
    for (int nt = 0; nt < 8; ++nt) {
        int t = t0 + nt * 16 + n;
        if (t >= Tout) continue;
        unsigned lo = 0, hi = 0;
#pragma unroll
        for (int r = 0; r < 8; ++r) {
            float y = (float)acc[nt][r] * Ar[r] + Cr[r];
            if (si_relu >= 0) {
                int qr = clampi(q_rne(fmaxf(y, 0.0f) / s_relu), 0, 255);
                y = (float)qr * s_relu;
            }
            int q = clampi(q_rne(y / s_next), -128, 127);
            unsigned byte = (unsigned)(q & 0xff);
            if (r < 4) lo |= byte << (8 * r);
            else       hi |= byte << (8 * (r - 4));
        }
        *(int2*)(out + ((size_t)b * Tout + t) * 128 + wave * 16 + half * 8) =
            make_int2((int)lo, (int)hi);
    }
}

__launch_bounds__(256, 2)
__global__ void fc_i8_wmma_kernel(const signed char* __restrict__ in, int Tbuf,
                                  const int* __restrict__ gather,
                                  const int* __restrict__ wq,
                                  const float* __restrict__ act_scales,
                                  const float* __restrict__ w_scales,
                                  float* __restrict__ out)
{
    __shared__ __align__(16) signed char lds_act[128 * 128];
    const int Cin = 128, Tout = 1536;
    const int tid  = threadIdx.x;
    const int lane = tid & 31;
    const int wave = tid >> 5;
    const int b    = blockIdx.y;
    const int t0   = blockIdx.x * 128;

    for (int u = tid; u < 128 * 8; u += 256) {
        int row = u >> 3;
        int off = u & 7;
        int tg = gather[t0 + row];
        const int4* src = (const int4*)(in + ((size_t)b * Tbuf + tg) * Cin) + off;
        *((int4*)(lds_act + row * Cin) + off) = *src;
    }
    __syncthreads();

    const int n    = lane & 15;
    const int half = lane >> 4;
    v8i acc = v8i{};
#pragma unroll
    for (int ks = 0; ks < 2; ++ks) {
        const int4* wp = (const int4*)wq + ((ks * 32 + lane) << 1);
        int4 a0 = wp[0], a1 = wp[1];
        v8i a;
        a[0] = a0.x; a[1] = a0.y; a[2] = a0.z; a[3] = a0.w;
        a[4] = a1.x; a[5] = a1.y; a[6] = a1.z; a[7] = a1.w;
        const signed char* bp =
            lds_act + (size_t)(wave * 16 + n) * Cin + ks * 64 + half * 16;
        int4 blo = *(const int4*)bp;
        int4 bhi = *(const int4*)(bp + 32);
        v8i bf;
        bf[0] = blo.x; bf[1] = blo.y; bf[2] = blo.z; bf[3] = blo.w;
        bf[4] = bhi.x; bf[5] = bhi.y; bf[6] = bhi.z; bf[7] = bhi.w;
        acc = __builtin_amdgcn_wmma_i32_16x16x64_iu8(
            true, a, true, bf, acc, false, false);
    }

    float deq = act_scales[10] * w_scales[6];
    float s11 = act_scales[11];
    int t = t0 + wave * 16 + n;
#pragma unroll
    for (int r = 0; r < 8; ++r) {
        int co = half * 8 + r;
        if (co < 10) {
            float y = (float)acc[r] * deq;
            int q = clampi(q_rne(y / s11), -128, 127);
            y = (float)q * s11;
            y = fminf(fmaxf(y, 0.0f), 127.0f);
            out[((size_t)b * 10 + co) * Tout + t] = y;
        }
    }
}

extern "C" void kernel_launch(void* const* d_in, const int* in_sizes, int n_in,
                              void* d_out, int out_size, void* d_ws, size_t ws_size,
                              hipStream_t stream)
{
    (void)in_sizes; (void)n_in; (void)out_size; (void)ws_size;
    const float* x      = (const float*)d_in[0];
    const float* w0c1   = (const float*)d_in[1];
    const float* w0c2   = (const float*)d_in[2];
    const float* w1c1   = (const float*)d_in[3];
    const float* w1c2   = (const float*)d_in[4];
    const float* w2c1   = (const float*)d_in[5];
    const float* w2c2   = (const float*)d_in[6];
    const float* fc_w   = (const float*)d_in[7];
    const float* bn     = (const float*)d_in[8];
    const float* ascal  = (const float*)d_in[9];
    const float* wscal  = (const float*)d_in[10];
    const int* idx_b0c1 = (const int*)d_in[12];
    const int* idx_b0c2 = (const int*)d_in[13];
    const int* idx_b1c1 = (const int*)d_in[14];
    const int* idx_b1c2 = (const int*)d_in[15];
    const int* idx_b2c1 = (const int*)d_in[16];
    const int* idx_b2c2 = (const int*)d_in[17];
    const int* idx_in   = (const int*)d_in[11];

    signed char* ws8 = (signed char*)d_ws;
    const size_t WO[7] = {0, 24576, 73728, 122880, 172032, 221184, 270336};
    signed char* act0 = ws8 + 272384;
    signed char* P0   = act0 + (size_t)32 * 12288 * 64;
    signed char* P1   = P0 + (size_t)32 * 12286 * 128;

    pack_weights_kernel<<<24, 256, 0, stream>>>(w0c1, wscal, 0,  64, 3, 128, 3, 8, (int*)(ws8 + WO[0]));
    pack_weights_kernel<<<48, 256, 0, stream>>>(w0c2, wscal, 1, 128, 3, 128, 6, 8, (int*)(ws8 + WO[1]));
    pack_weights_kernel<<<48, 256, 0, stream>>>(w1c1, wscal, 2, 128, 3, 128, 6, 8, (int*)(ws8 + WO[2]));
    pack_weights_kernel<<<48, 256, 0, stream>>>(w1c2, wscal, 3, 128, 3, 128, 6, 8, (int*)(ws8 + WO[3]));
    pack_weights_kernel<<<48, 256, 0, stream>>>(w2c1, wscal, 4, 128, 3, 128, 6, 8, (int*)(ws8 + WO[4]));
    pack_weights_kernel<<<48, 256, 0, stream>>>(w2c2, wscal, 5, 128, 3, 128, 6, 8, (int*)(ws8 + WO[5]));
    pack_weights_kernel<<< 2, 256, 0, stream>>>(fc_w, wscal, 6, 128, 1,  10, 2, 1, (int*)(ws8 + WO[6]));

    prep_kernel<<<98304, 256, 0, stream>>>(x, idx_in, ascal, act0);

    conv_i8_wmma_kernel<<<dim3(96, 32), 256, 0, stream>>>(
        act0, 12288, nullptr, 12288, 64, 3, (int*)(ws8 + WO[0]),
        ascal, 1, -1, 2, wscal, 0, bn, 0 * 512, P0, 12286);
    conv_i8_wmma_kernel<<<dim3(64, 32), 256, 0, stream>>>(
        P0, 12286, idx_b0c1, 8192, 128, 6, (int*)(ws8 + WO[1]),
        ascal, 2, 3, 4, wscal, 1, bn, 1 * 512, P1, 8190);
    conv_i8_wmma_kernel<<<dim3(48, 32), 256, 0, stream>>>(
        P1, 8190, idx_b0c2, 6144, 128, 6, (int*)(ws8 + WO[2]),
        ascal, 4, -1, 5, wscal, 2, bn, 2 * 512, P0, 6142);
    conv_i8_wmma_kernel<<<dim3(32, 32), 256, 0, stream>>>(
        P0, 6142, idx_b1c1, 4096, 128, 6, (int*)(ws8 + WO[3]),
        ascal, 5, 6, 7, wscal, 3, bn, 3 * 512, P1, 4094);
    conv_i8_wmma_kernel<<<dim3(24, 32), 256, 0, stream>>>(
        P1, 4094, idx_b1c2, 3072, 128, 6, (int*)(ws8 + WO[4]),
        ascal, 7, -1, 8, wscal, 4, bn, 4 * 512, P0, 3070);
    conv_i8_wmma_kernel<<<dim3(16, 32), 256, 0, stream>>>(
        P0, 3070, idx_b2c1, 2048, 128, 6, (int*)(ws8 + WO[5]),
        ascal, 8, 9, 10, wscal, 5, bn, 5 * 512, P1, 2046);
    fc_i8_wmma_kernel<<<dim3(12, 32), 256, 0, stream>>>(
        P1, 2046, idx_b2c2, (int*)(ws8 + WO[6]), ascal, wscal, (float*)d_out);
}
